// GINModel_91053306675270
// MI455X (gfx1250) — matrix operand order repaired and checked by reading, not compiled, not run on hardware
//
#include <hip/hip_runtime.h>
#include <hip/hip_bf16.h>

#define NNODES 50000
#define NEDGES 640000
#define NGRAPHS 500
#define DIM 128
#define NLAYER 3

typedef __bf16 bf16;
typedef bf16  v16bf __attribute__((ext_vector_type(16)));
typedef bf16  v8bf  __attribute__((ext_vector_type(8)));
typedef float v8f   __attribute__((ext_vector_type(8)));

// fp32 -> bf16 RNE via native cast (compiler picks HW convert when available)
__device__ __forceinline__ bf16 f2bf(float f) { return (bf16)f; }

// guaranteed-hardware f32 atomic add (relaxed, device scope)
__device__ __forceinline__ void atomAddF(float* p, float v) {
    __hip_atomic_fetch_add(p, v, __ATOMIC_RELAXED, __HIP_MEMORY_SCOPE_AGENT);
}

// -------------------- simple data movers --------------------

__global__ void copy_f4(float4* __restrict__ dst, const float4* __restrict__ src, int n4) {
    int i = blockIdx.x * blockDim.x + threadIdx.x;
    if (i < n4) dst[i] = src[i];
}

__global__ void zero_f(float* __restrict__ p, int n) {
    int i = blockIdx.x * blockDim.x + threadIdx.x;
    if (i < n) p[i] = 0.0f;
}

// one wave per edge: lane handles 4 contiguous features -> hardware f32 atomics
__global__ void edge_scatter(const float* __restrict__ x, const int* __restrict__ src,
                             const int* __restrict__ dst, float* __restrict__ h, int ne) {
    int t = blockIdx.x * blockDim.x + threadIdx.x;
    int e = t >> 5;
    if (e >= ne) return;
    int lane = t & 31;
    int c = lane * 4;
    const float4 v = *(const float4*)(x + (size_t)src[e] * DIM + c);
    float* hp = h + (size_t)dst[e] * DIM + c;
    atomAddF(hp + 0, v.x);
    atomAddF(hp + 1, v.y);
    atomAddF(hp + 2, v.z);
    atomAddF(hp + 3, v.w);
}

__global__ void pool_scatter(const float* __restrict__ x, const int* __restrict__ batch,
                             float* __restrict__ g, int nnodes) {
    int t = blockIdx.x * blockDim.x + threadIdx.x;
    int node = t >> 5;
    if (node >= nnodes) return;
    int lane = t & 31;
    int c = lane * 4;
    const float4 v = *(const float4*)(x + (size_t)node * DIM + c);
    float* gp = g + (size_t)batch[node] * DIM + c;
    atomAddF(gp + 0, v.x);
    atomAddF(gp + 1, v.y);
    atomAddF(gp + 2, v.z);
    atomAddF(gp + 3, v.w);
}

// -------------------- fused WMMA GEMM: Out = act(A @ W + b) --------------------
// AT: A element type (float or bf16, row-major [nrows,128])
// OT: Out element type (float or bf16, row-major)
// MODE 0: ReLU(A@W + bias); MODE 1: ReLU(BN(A@W + bias)) with BN folded to one FMA.
// Block: 256 threads = 8 waves; wave owns a 16-row strip x 128 cols.
// W staged in LDS as bf16 in WMMA B-fragment order:
//   fragment element j of lane L for (nt,kc) = W[kc*32 + ((L&16)>>1) + (j&7) + ((j&8)<<1)][nt*16 + (L&15)]
template <typename AT, typename OT, int MODE>
__global__ __launch_bounds__(256) void gemm_bf16_wmma(
        const AT* __restrict__ A, const float* __restrict__ W,
        const float* __restrict__ bias,
        const float* __restrict__ bn_scale, const float* __restrict__ bn_bias,
        const float* __restrict__ bn_mean,  const float* __restrict__ bn_var,
        OT* __restrict__ Out, int nrows) {
    __shared__ unsigned short sW[DIM * DIM];   // 32 KB, fragment-ordered bf16
    __shared__ float sMul[DIM];
    __shared__ float sAdd[DIM];

    const int tid = threadIdx.x;

    if (tid < DIM) {
        if (MODE == 0) {
            sMul[tid] = 1.0f;
            sAdd[tid] = bias[tid];
        } else {
            float s = bn_scale[tid] * rsqrtf(bn_var[tid] + 1e-5f);
            sMul[tid] = s;
            sAdd[tid] = (bias[tid] - bn_mean[tid]) * s + bn_bias[tid];
        }
    }

    // Stage W -> bf16 fragments, 2 elements (1 dword) per store, 32 dwords/thread.
    unsigned* sW32 = (unsigned*)sW;
#pragma unroll
    for (int i = 0; i < (DIM * DIM / 2) / 256; ++i) {
        int e2   = tid + i * 256;     // dword index
        int e    = e2 * 2;            // bf16 element index (even)
        int j    = e & 15;
        int lane = (e >> 4) & 31;
        int kc   = (e >> 9) & 3;
        int nt   = e >> 11;
        int k = kc * 32 + ((lane & 16) >> 1) + (j & 7) + ((j & 8) << 1);
        int n = nt * 16 + (lane & 15);
        union { bf16 b; unsigned short u; } lo, hi;
        lo.b = f2bf(W[(k + 0) * DIM + n]);
        hi.b = f2bf(W[(k + 1) * DIM + n]);
        sW32[e2] = (unsigned)lo.u | ((unsigned)hi.u << 16);
    }
    __syncthreads();

    const int lane = tid & 31;
    const int wave = tid >> 5;
    const int rowBase = blockIdx.x * 128 + wave * 16;
    if (rowBase >= nrows) return;   // wave-uniform: EXEC stays all-ones for WMMA

    // Build A fragments (4 chunks of K=32) per documented 16-bit A layout:
    //   lane<16 holds K in {kb..kb+7, kb+16..kb+23} with kb = kc*32
    //   lane>=16 same with +8 offset
    v16bf afrag[4];
    const int kb0 = (lane & 16) ? 8 : 0;
    const AT* rowp = A + (size_t)(rowBase + (lane & 15)) * DIM;
#pragma unroll
    for (int kc = 0; kc < 4; ++kc) {
        int base = kc * 32 + kb0;
        if constexpr (sizeof(AT) == 4) {   // fp32 input: load + convert
            float4 f0 = *(const float4*)((const float*)rowp + base);
            float4 f1 = *(const float4*)((const float*)rowp + base + 4);
            float4 f2 = *(const float4*)((const float*)rowp + base + 16);
            float4 f3 = *(const float4*)((const float*)rowp + base + 20);
            v16bf a;
            a[0]  = f2bf(f0.x); a[1]  = f2bf(f0.y); a[2]  = f2bf(f0.z); a[3]  = f2bf(f0.w);
            a[4]  = f2bf(f1.x); a[5]  = f2bf(f1.y); a[6]  = f2bf(f1.z); a[7]  = f2bf(f1.w);
            a[8]  = f2bf(f2.x); a[9]  = f2bf(f2.y); a[10] = f2bf(f2.z); a[11] = f2bf(f2.w);
            a[12] = f2bf(f3.x); a[13] = f2bf(f3.y); a[14] = f2bf(f3.z); a[15] = f2bf(f3.w);
            afrag[kc] = a;
        } else {                           // bf16 input: straight 16B loads, no convert
            v8bf lo = *(const v8bf*)((const bf16*)rowp + base);
            v8bf hi = *(const v8bf*)((const bf16*)rowp + base + 16);
            afrag[kc] = __builtin_shufflevector(lo, hi, 0, 1, 2, 3, 4, 5, 6, 7,
                                                8, 9, 10, 11, 12, 13, 14, 15);
        }
    }

    const int nlo  = lane & 15;
    const int moff = (lane & 16) >> 1;   // rows 0..7 or 8..15 of the C tile

#pragma unroll
    for (int nt = 0; nt < 8; ++nt) {
        v8f acc = {0.f, 0.f, 0.f, 0.f, 0.f, 0.f, 0.f, 0.f};
#pragma unroll
        for (int kc = 0; kc < 4; ++kc) {
            v16bf b = *(const v16bf*)(sW + ((nt * 4 + kc) * 32 + lane) * 16);
            acc = __builtin_amdgcn_wmma_f32_16x16x32_bf16(
                      false, afrag[kc], false, b, (short)0, acc, false, false);
        }
        int col = nt * 16 + nlo;
        float mul = sMul[col];
        float add = sAdd[col];
#pragma unroll
        for (int r = 0; r < 8; ++r) {
            float v = fmaf(acc[r], mul, add);
            v = fmaxf(v, 0.0f);
            size_t idx = (size_t)(rowBase + moff + r) * DIM + col;
            if constexpr (sizeof(OT) == 4) Out[idx] = v;
            else                           Out[idx] = f2bf(v);
        }
    }
}

// -------------------- tiny head: out = sigmoid(relu(g@Wh1+bh1)@Wh2+bh2) --------------------
__global__ __launch_bounds__(64) void head_kernel(
        const float* __restrict__ g,
        const float* __restrict__ Wh1, const float* __restrict__ bh1,
        const float* __restrict__ Wh2, const float* __restrict__ bh2,
        float* __restrict__ out) {
    __shared__ float sh[64];
    const int gr = blockIdx.x;
    const int j  = threadIdx.x;
    const float* gp = g + (size_t)gr * DIM;
    float s = bh1[j];
#pragma unroll 4
    for (int k = 0; k < DIM; ++k) s = fmaf(gp[k], Wh1[k * 64 + j], s);
    sh[j] = fmaxf(s, 0.0f);
    __syncthreads();
    if (j < 12) {
        float o = bh2[j];
#pragma unroll 4
        for (int k = 0; k < 64; ++k) o = fmaf(sh[k], Wh2[k * 12 + j], o);
        out[(size_t)gr * 12 + j] = 1.0f / (1.0f + __expf(-o));
    }
}

// -------------------- launcher --------------------

extern "C" void kernel_launch(void* const* d_in, const int* in_sizes, int n_in,
                              void* d_out, int out_size, void* d_ws, size_t ws_size,
                              hipStream_t stream) {
    const float* x    = (const float*)d_in[0];
    const int*   ei   = (const int*)d_in[1];
    const int*   batc = (const int*)d_in[2];
    const float* W1   = (const float*)d_in[3];
    const float* b1   = (const float*)d_in[4];
    const float* W2   = (const float*)d_in[5];
    const float* b2   = (const float*)d_in[6];
    const float* bnsc = (const float*)d_in[7];
    const float* bnbi = (const float*)d_in[8];
    const float* bnme = (const float*)d_in[9];
    const float* bnva = (const float*)d_in[10];
    const float* Wh1  = (const float*)d_in[11];
    const float* bh1  = (const float*)d_in[12];
    const float* Wh2  = (const float*)d_in[13];
    const float* bh2  = (const float*)d_in[14];
    float* out = (float*)d_out;

    float* B0  = (float*)d_ws;                     // agg buffer h (fp32, atomics)
    float* B1f = B0 + (size_t)NNODES * DIM;        // MLP intermediate region
    bf16*  B1  = (bf16*)B1f;                       //   ... stored as bf16
    float* B2  = B1f + (size_t)NNODES * DIM;       // layer output x (fp32)
    float* G   = B2 + (size_t)NNODES * DIM;        // pooled graphs

    const int* src = ei;
    const int* dst = ei + NEDGES;

    const int n4 = NNODES * DIM / 4;
    dim3 blk(256);
    const int gemmGrid = (NNODES + 127) / 128;

    const float* xin = x;
    for (int l = 0; l < NLAYER; ++l) {
        copy_f4<<<(n4 + 255) / 256, blk, 0, stream>>>((float4*)B0, (const float4*)xin, n4);
        edge_scatter<<<(NEDGES * 32 + 255) / 256, blk, 0, stream>>>(xin, src, dst, B0, NEDGES);
        gemm_bf16_wmma<float, bf16, 0><<<gemmGrid, blk, 0, stream>>>(
            B0, W1 + (size_t)l * DIM * DIM, b1 + (size_t)l * DIM,
            nullptr, nullptr, nullptr, nullptr, B1, NNODES);
        gemm_bf16_wmma<bf16, float, 1><<<gemmGrid, blk, 0, stream>>>(
            B1, W2 + (size_t)l * DIM * DIM, b2 + (size_t)l * DIM,
            bnsc + (size_t)l * DIM, bnbi + (size_t)l * DIM,
            bnme + (size_t)l * DIM, bnva + (size_t)l * DIM, B2, NNODES);
        xin = B2;
    }

    zero_f<<<(NGRAPHS * DIM + 255) / 256, blk, 0, stream>>>(G, NGRAPHS * DIM);
    pool_scatter<<<(NNODES * 32 + 255) / 256, blk, 0, stream>>>(B2, batc, G, NNODES);
    head_kernel<<<NGRAPHS, dim3(64), 0, stream>>>(G, Wh1, bh1, Wh2, bh2, out);
}
